// RMSPELoss_83700322664468
// MI455X (gfx1250) — matrix operand order repaired and checked by reading, not compiled, not run on hardware
//
#include <hip/hip_runtime.h>
#include <math.h>

// RMSPE loss with optimal permutation assignment (N=5 sources, brute-force 120 perms)
// einsum('bij,pij->bp') mapped onto V_WMMA_F32_16X16X32_F16:
//   A = 16 batch rows x 32 (25 cost entries + 7 zero pad), f16
//   B = 32 x 16 one-hot permutation tile (8 tiles cover 120 perms, padded cols
//       duplicate perm 119 so they never win the min), f16 (exact)
//   D = 16x16 f32 permutation costs, min-reduced across tiles + lanes.

typedef __attribute__((ext_vector_type(16))) _Float16 v16h;
typedef __attribute__((ext_vector_type(8)))  float    v8f;

#define NSRC    5
#define NPERM   120
#define NBATCH  524288
#define TPB     256
#define NBLK    512
#define NWAVES  ((NBLK * TPB) / 32)    /* 4096  */
#define NTILES  (NBATCH / 16)          /* 32768; 32768 % 4096 == 0 -> uniform trips */

// ((x + pi/2) mod pi) - pi/2, matching jnp semantics (mod result in [0, pi))
__device__ __forceinline__ float wrap_pi(float x) {
  const float PI  = 3.14159265358979323846f;
  const float HPI = 1.57079632679489661923f;
  return x - PI * floorf((x + HPI) * (1.0f / PI));
}

__global__ __launch_bounds__(TPB) void rmspe_stage1(
    const float* __restrict__ pred,
    const float* __restrict__ tgt,
    float* __restrict__ partials)
{
  const int lane = threadIdx.x & 31;
  const int hf   = lane >> 4;     // 0: lanes 0-15, 1: lanes 16-31
  const int sub  = lane & 15;
  const int wid  = (blockIdx.x * TPB + threadIdx.x) >> 5;  // wave-uniform

  // ---- Build the 8 B fragments (one-hot perm tiles) once per wave ----
  // B (32x16) k order chosen by this code (A matches): VGPR v, lanes 0-15: N=lane, K={2v,2v+1};
  // lanes 16-31: N=lane-16, K={16+2v,17+2v}.
  v16h Bf[8];
  #pragma unroll
  for (int t = 0; t < 8; ++t) {
    int p = t * 16 + sub;
    if (p >= NPERM) p = NPERM - 1;          // duplicate last perm in pad columns
    // lexicographic (itertools.permutations) decode via factorial number system
    int rem[5] = {0, 1, 2, 3, 4};
    int sig[5];
    int r = p;
    const int fact[4] = {24, 6, 2, 1};
    #pragma unroll
    for (int i = 0; i < 4; ++i) {
      int q = r / fact[i];
      r -= q * fact[i];
      sig[i] = rem[q];
      #pragma unroll
      for (int j = 0; j < 4; ++j)
        if (j >= q) rem[j] = rem[j + 1];
    }
    sig[4] = rem[0];
    #pragma unroll
    for (int v = 0; v < 8; ++v) {
      #pragma unroll
      for (int e = 0; e < 2; ++e) {
        int k = 2 * v + e + 16 * hf;
        float b = 0.0f;
        if (k < NSRC * NSRC) {
          int i = k / NSRC, j = k - NSRC * i;
          b = (sig[i] == j) ? 1.0f : 0.0f;  // PERM_ONEHOT[p,i,j]
        }
        Bf[t][2 * v + e] = (_Float16)b;
      }
    }
  }

  float acc = 0.0f;
  for (int tile = wid; tile < NTILES; tile += NWAVES) {
    const int row = tile * 16 + sub;        // lanes L and L+16 share row M=L
    const float* pr = pred + row * NSRC;
    const float* tg = tgt  + row * NSRC;
    float P[NSRC], T[NSRC];
    #pragma unroll
    for (int i = 0; i < NSRC; ++i) { P[i] = pr[i]; T[i] = tg[i]; }

    if (tile + NWAVES < NTILES) {           // wave-uniform guard
      __builtin_prefetch(pred + (row + NWAVES * 16) * NSRC, 0, 1);
      __builtin_prefetch(tgt  + (row + NWAVES * 16) * NSRC, 0, 1);
    }

    // A fragment (16x32 f16): lane L<16 -> K {0..15}; lane L+16 -> K {16..31}
    v16h a;
    #pragma unroll
    for (int v = 0; v < 8; ++v) {
      int kb = 2 * v + 16 * hf;
      #pragma unroll
      for (int e = 0; e < 2; ++e) {
        int k = kb + e;
        float c = 0.0f;
        if (k < NSRC * NSRC) {
          int i = k / NSRC, j = k - NSRC * i;
          float dd = wrap_pi(P[i] - T[j]);  // diff[i][j] = pred_i - tgt_j, wrapped
          c = dd * dd;
        }
        a[2 * v + e] = (_Float16)c;
      }
    }

    // 8 perm-tiles: D = A x B_t, running elementwise min per lane
    float m[8];
    #pragma unroll
    for (int e = 0; e < 8; ++e) m[e] = 3.0e38f;
    #pragma unroll
    for (int t = 0; t < 8; ++t) {
      v8f cz = {};
      v8f d = __builtin_amdgcn_wmma_f32_16x16x32_f16(
          false, a, false, Bf[t], (short)0, cz, false, false);
      #pragma unroll
      for (int e = 0; e < 8; ++e) m[e] = fminf(m[e], d[e]);
    }

    // Cross-lane min within each 16-lane half (D: VGPR e holds row M=e (half 0)
    // or M=e+8 (half 1), N = lane%16). Then rmspe = sqrt(min/5), summed.
    float s = 0.0f;
    #pragma unroll
    for (int e = 0; e < 8; ++e) {
      float x = m[e];
      x = fminf(x, __shfl_xor(x, 1));
      x = fminf(x, __shfl_xor(x, 2));
      x = fminf(x, __shfl_xor(x, 4));
      x = fminf(x, __shfl_xor(x, 8));
      s += sqrtf(x * (1.0f / NSRC));
    }
    acc += s;   // half 0: rows 0-7 of the tile; half 1: rows 8-15
  }

  // Combine the two halves; one partial per wave (deterministic stage-2 reduce)
  float other = __shfl_xor(acc, 16);
  if (lane == 0) partials[wid] = acc + other;
}

__global__ __launch_bounds__(TPB) void rmspe_stage2(
    const float* __restrict__ partials, float* __restrict__ out)
{
  __shared__ float sm[TPB];
  float s = 0.0f;
  for (int i = threadIdx.x; i < NWAVES; i += TPB) s += partials[i];  // fixed order
  sm[threadIdx.x] = s;
  __syncthreads();
  for (int off = TPB / 2; off > 0; off >>= 1) {
    if (threadIdx.x < off) sm[threadIdx.x] += sm[threadIdx.x + off];
    __syncthreads();
  }
  if (threadIdx.x == 0) out[0] = sm[0];
}

extern "C" void kernel_launch(void* const* d_in, const int* in_sizes, int n_in,
                              void* d_out, int out_size, void* d_ws, size_t ws_size,
                              hipStream_t stream) {
  const float* pred = (const float*)d_in[0];   // [524288, 5] f32
  const float* tgt  = (const float*)d_in[1];   // [524288, 5] f32
  float* out      = (float*)d_out;             // [1] f32
  float* partials = (float*)d_ws;              // NWAVES floats of scratch

  rmspe_stage1<<<NBLK, TPB, 0, stream>>>(pred, tgt, partials);
  rmspe_stage2<<<1, TPB, 0, stream>>>(partials, out);
}
